// LSTMBlock_77988016160946
// MI455X (gfx1250) — matrix operand order repaired and checked by reading, not compiled, not run on hardware
//
#include <hip/hip_runtime.h>
#include <hip/hip_bf16.h>

// ---------------------------------------------------------------------------
// LSTM cell, MI455X (gfx1250, wave32, WMMA, async global->LDS).
// Pass 1: convert x|h and [Wi|Wf|Wg|Wo] (input+hidden) to bf16 hi/lo planes
//         in workspace (each element converted exactly once).
// Pass 2: GEMM gates = A' @ W'^T with split-bf16 WMMA (hi*hi + lo*hi + hi*lo),
//         tiles streamed with GLOBAL_LOAD_ASYNC_TO_LDS_B128 (ASYNCcnt),
//         double-buffered 2x32KB LDS, XOR-swizzled chunks (no padding),
//         fused sigmoid/tanh/c_new/h_new epilogue per-wave (all 4 gates local).
// Fallback: single-pass fused kernel (round-1) if workspace too small.
// ---------------------------------------------------------------------------

typedef __attribute__((ext_vector_type(16))) __bf16 v16bf;
typedef __attribute__((ext_vector_type(8)))  __bf16 v8bf;
typedef __attribute__((ext_vector_type(8)))  float  v8f;

#define B_DIM 4096
#define K_DIM 1024            // IN == H == 1024
#define H_DIM 1024
#define KW    2048            // concatenated K (x-phase + h-phase)
#define BM    128             // batch rows per block
#define BN    32              // hidden cols per block (per gate)
#define BK    32              // K per stage
#define NITER (KW / BK)       // 64

#define LDS_AS __attribute__((address_space(3)))

__device__ __forceinline__ float sigmoid_f(float x) {
  return 1.0f / (1.0f + __expf(-x));
}
__device__ __forceinline__ float tanh_f(float x) {
  float e = __expf(-2.0f * fabsf(x));
  float t = (1.0f - e) / (1.0f + e);
  return copysignf(t, x);
}

union V16 { v16bf v; v8bf h[2]; };

__device__ __forceinline__ void cvt8(const float* f, v8bf* hi, v8bf* lo) {
#pragma unroll
  for (int j = 0; j < 8; ++j) {
    __bf16 hb = (__bf16)f[j];
    (*hi)[j] = hb;
    (*lo)[j] = (__bf16)(f[j] - (float)hb);
  }
}

// ---------------------------------------------------------------------------
// Pass 1: f32 -> (bf16 hi, bf16 lo) planes, concatenated K layout [row][2048]
// ---------------------------------------------------------------------------
struct CvtParams {
  const float* A[2];        // x, h_t                [4096,1024]
  const float* W[2][4];     // [phase][gate]         [1024,1024]
  __bf16 *Ahi, *Alo, *Whi, *Wlo;   // each [4096,2048]
};

__global__ __launch_bounds__(256) void lstm_convert_kernel(CvtParams C) {
  const long e0  = ((long)blockIdx.x * 256 + threadIdx.x) * 8;  // elem in [0, 8M)
  const int row  = (int)(e0 >> 11);        // /2048
  const int k    = (int)(e0 & 2047);
  const int ph   = k >> 10;
  const int kk   = k & 1023;

  const float* src;
  __bf16 *dhi, *dlo;
  if (blockIdx.y == 0) {                    // A' = [x | h]
    src = C.A[ph] + (size_t)row * K_DIM + kk;
    dhi = C.Ahi;  dlo = C.Alo;
  } else {                                  // W' = gates stacked, K concat
    const int g = row >> 10, n = row & 1023;
    src = C.W[ph][g] + (size_t)n * K_DIM + kk;
    dhi = C.Whi;  dlo = C.Wlo;
  }
  float4 f0 = ((const float4*)src)[0];
  float4 f1 = ((const float4*)src)[1];
  float f[8] = { f0.x, f0.y, f0.z, f0.w, f1.x, f1.y, f1.z, f1.w };
  v8bf hi, lo;
  cvt8(f, &hi, &lo);
  const size_t o = (size_t)row * KW + k;
  *(v8bf*)(dhi + o) = hi;
  *(v8bf*)(dlo + o) = lo;
}

// ---------------------------------------------------------------------------
// Pass 2: async-WMMA GEMM + fused LSTM epilogue
// ---------------------------------------------------------------------------
struct GemmParams {
  const __bf16 *Ahi, *Alo, *Whi, *Wlo;
  const float* bI[4];
  const float* bH[4];
  const float* cIn;
  float *hOut0, *hOut1, *cOut;
};

__device__ __forceinline__ void async_cp_b128(unsigned lds_byte, unsigned goff,
                                              const __bf16* base) {
  // GVS mode: LDS byte address in the VDST vgpr; global address is SADDR base plus u32 VADDR offset
  asm volatile("global_load_async_to_lds_b128 %0, %1, %2"
               :: "v"(lds_byte), "v"(goff), "s"(base) : "memory");
}

__device__ __forceinline__ void wait_async0() {
#if __has_builtin(__builtin_amdgcn_s_wait_asynccnt)
  __builtin_amdgcn_s_wait_asynccnt(0);
#else
  asm volatile("s_wait_asynccnt 0x0" ::: "memory");
#endif
}

__global__ __launch_bounds__(256) void lstm_wmma_async_kernel(GemmParams P) {
  // 2 stages x 4 planes (Ahi,Alo,Whi,Wlo) x 128 rows x 64B = 65536 B
  __shared__ __bf16 smem[2 * 4 * BM * BK];

  const int t    = threadIdx.x;
  const int wave = t >> 5;
  const int lane = t & 31;
  const int kh   = lane >> 4;
  const int l16  = lane & 15;

  const int rowBase = blockIdx.y * BM;
  const int n0      = blockIdx.x * BN;

  // raw LDS byte offset of smem (addrspace(3) pointer value)
  unsigned smemBase;
  {
    LDS_AS __bf16* p3 = (LDS_AS __bf16*)smem;
    smemBase = (unsigned)(uintptr_t)p3;
  }
  const char* smemc = (const char*)smem;

  // ---- async staging map: thread t copies row tr, 16B chunks {cp, cp+1}
  const int tr = t >> 1;              // 0..127 tile row
  const int cp = (t & 1) * 2;         // chunk pair base 0 or 2
  const int sw = (tr >> 2) & 3;       // XOR swizzle for this row
  const unsigned ldsRow = (unsigned)tr * 64u;
  const unsigned ldsC0  = ldsRow + (unsigned)((cp     ^ sw) * 16);
  const unsigned ldsC1  = ldsRow + (unsigned)(((cp+1) ^ sw) * 16);
  // global byte offsets (row-major bf16, 4096 B per row), +kt*64 at issue
  const unsigned gA0 = (unsigned)(rowBase + tr) * (KW * 2) + cp * 16;
  const int wg = tr >> 5, wj = tr & 31;
  const unsigned gW0 = (unsigned)(wg * 1024 + n0 + wj) * (KW * 2) + cp * 16;

  auto issue_stage = [&](int kt, int stage) {
    const unsigned kB = (unsigned)kt * 64u;
    const unsigned sB = smemBase + (unsigned)stage * 32768u;
    async_cp_b128(sB +     0 + ldsC0, gA0 + kB,      P.Ahi);
    async_cp_b128(sB +     0 + ldsC1, gA0 + kB + 16, P.Ahi);
    async_cp_b128(sB +  8192 + ldsC0, gA0 + kB,      P.Alo);
    async_cp_b128(sB +  8192 + ldsC1, gA0 + kB + 16, P.Alo);
    async_cp_b128(sB + 16384 + ldsC0, gW0 + kB,      P.Whi);
    async_cp_b128(sB + 16384 + ldsC1, gW0 + kB + 16, P.Whi);
    async_cp_b128(sB + 24576 + ldsC0, gW0 + kB,      P.Wlo);
    async_cp_b128(sB + 24576 + ldsC1, gW0 + kB + 16, P.Wlo);
  };

  auto ld8 = [&](unsigned byteOff) -> v8bf {
    return *(const v8bf*)(smemc + byteOff);
  };

  v8f acc[4][2];
#pragma unroll
  for (int g = 0; g < 4; ++g)
#pragma unroll
    for (int nt = 0; nt < 2; ++nt)
      acc[g][nt] = (v8f){};

  // fragment read offsets (stage-relative, swizzled)
  const int ar = wave * 16 + l16;             // A tile row for this lane
  const int as = (ar >> 2) & 3;
  const unsigned aOffC0 = (unsigned)ar * 64u + (unsigned)(((kh)     ^ as) * 16); // K chunk kh
  const unsigned aOffC1 = (unsigned)ar * 64u + (unsigned)(((2 + kh) ^ as) * 16); // K chunk 2+kh

  issue_stage(0, 0);
  wait_async0();
  __syncthreads();

  for (int kt = 0; kt < NITER; ++kt) {
    const int cur = kt & 1;
    const unsigned sB = (unsigned)cur * 32768u;
    if (kt + 1 < NITER) issue_stage(kt + 1, cur ^ 1);   // overlap with WMMAs

    V16 ahi, alo;
    ahi.h[0] = ld8(sB + aOffC0);
    ahi.h[1] = ld8(sB + aOffC1);
    alo.h[0] = ld8(sB + 8192 + aOffC0);
    alo.h[1] = ld8(sB + 8192 + aOffC1);

#pragma unroll
    for (int g = 0; g < 4; ++g) {
#pragma unroll
      for (int nt = 0; nt < 2; ++nt) {
        const int br = g * 32 + nt * 16 + l16;          // W tile row = B column
        const int bs = (br >> 2) & 3;
        const unsigned bOff = (unsigned)br * 64u;
        const unsigned c0 = bOff + (unsigned)(((kh)     ^ bs) * 16);
        const unsigned c1 = bOff + (unsigned)(((2 + kh) ^ bs) * 16);
        V16 bhi, blo;
        bhi.h[0] = ld8(sB + 16384 + c0);
        bhi.h[1] = ld8(sB + 16384 + c1);
        blo.h[0] = ld8(sB + 24576 + c0);
        blo.h[1] = ld8(sB + 24576 + c1);

        v8f a = acc[g][nt];
        a = __builtin_amdgcn_wmma_f32_16x16x32_bf16(false, ahi.v, false, bhi.v,
                                                    (short)0, a, false, false);
        a = __builtin_amdgcn_wmma_f32_16x16x32_bf16(false, alo.v, false, bhi.v,
                                                    (short)0, a, false, false);
        a = __builtin_amdgcn_wmma_f32_16x16x32_bf16(false, ahi.v, false, blo.v,
                                                    (short)0, a, false, false);
        acc[g][nt] = a;
      }
    }

    if (kt + 1 < NITER) wait_async0();
    __syncthreads();
  }

  // ---- fused LSTM epilogue (all 4 gates live in this wave's accumulators)
  float bias[4][2];
#pragma unroll
  for (int g = 0; g < 4; ++g)
#pragma unroll
    for (int nt = 0; nt < 2; ++nt) {
      const int hc = n0 + nt * 16 + l16;
      bias[g][nt] = P.bI[g][hc] + P.bH[g][hc];
    }

#pragma unroll
  for (int nt = 0; nt < 2; ++nt) {
    const int hc = n0 + nt * 16 + l16;
#pragma unroll
    for (int r = 0; r < 8; ++r) {
      const int mrow = rowBase + wave * 16 + r + kh * 8;
      const size_t idx = (size_t)mrow * H_DIM + hc;
      const float iv = sigmoid_f(acc[0][nt][r] + bias[0][nt]);
      const float fv = sigmoid_f(acc[1][nt][r] + bias[1][nt]);
      const float gv = tanh_f   (acc[2][nt][r] + bias[2][nt]);
      const float ov = sigmoid_f(acc[3][nt][r] + bias[3][nt]);
      const float cn = fv * P.cIn[idx] + iv * gv;
      const float hn = ov * tanh_f(cn);
      P.hOut0[idx] = hn;
      if (P.hOut1) P.hOut1[idx] = hn;
      if (P.cOut)  P.cOut[idx]  = cn;
    }
  }
}

// ---------------------------------------------------------------------------
// Fallback: single-pass fused kernel (round-1), used if ws small
// ---------------------------------------------------------------------------
#define LDSTR 40

struct LstmParams {
  const float* A[2];
  const float* W[2][4];
  const float* bI[4];
  const float* bH[4];
  const float* cIn;
  float *hOut0, *hOut1, *cOut;
};

__global__ __launch_bounds__(256) void lstm_wmma_fused_kernel(LstmParams P)
{
  __shared__ __bf16 sAhi[BM * LDSTR];
  __shared__ __bf16 sAlo[BM * LDSTR];
  __shared__ __bf16 sBhi[BM * LDSTR];
  __shared__ __bf16 sBlo[BM * LDSTR];

  const int t    = threadIdx.x;
  const int wave = t >> 5;
  const int lane = t & 31;
  const int kh   = lane >> 4;
  const int l16  = lane & 15;

  const int rowBase = blockIdx.y * BM;
  const int n0      = blockIdx.x * BN;

  const int aRow    = t >> 1;
  const int cOff    = (t & 1) * 16;
  const int bGate   = t >> 6;
  const int bRow    = (t >> 1) & 31;
  const int bLdsRow = bGate * 32 + bRow;

  float4 rA[4], rB[4];

  auto load_tile = [&](int kt) {
    const int kglob = kt * BK;
    const int phase = kglob >> 10;
    const int kk    = kglob & (K_DIM - 1);
    const float4* ap4 =
        (const float4*)(P.A[phase] + (size_t)(rowBase + aRow) * K_DIM + kk + cOff);
    rA[0] = ap4[0]; rA[1] = ap4[1]; rA[2] = ap4[2]; rA[3] = ap4[3];
    const float4* wp4 =
        (const float4*)(P.W[phase][bGate] + (size_t)(n0 + bRow) * K_DIM + kk + cOff);
    rB[0] = wp4[0]; rB[1] = wp4[1]; rB[2] = wp4[2]; rB[3] = wp4[3];
  };

  auto cvt_store8 = [&](float4 a, float4 b, __bf16* dhi, __bf16* dlo) {
    float f[8] = { a.x, a.y, a.z, a.w, b.x, b.y, b.z, b.w };
    v8bf hi, lo;
    cvt8(f, &hi, &lo);
    *(v8bf*)dhi = hi;
    *(v8bf*)dlo = lo;
  };

  auto store_tile = [&]() {
    const int abase = aRow * LDSTR + cOff;
    cvt_store8(rA[0], rA[1], &sAhi[abase],     &sAlo[abase]);
    cvt_store8(rA[2], rA[3], &sAhi[abase + 8], &sAlo[abase + 8]);
    const int bbase = bLdsRow * LDSTR + cOff;
    cvt_store8(rB[0], rB[1], &sBhi[bbase],     &sBlo[bbase]);
    cvt_store8(rB[2], rB[3], &sBhi[bbase + 8], &sBlo[bbase + 8]);
  };

  v8f acc[4][2];
#pragma unroll
  for (int g = 0; g < 4; ++g)
#pragma unroll
    for (int nt = 0; nt < 2; ++nt)
      acc[g][nt] = (v8f){};

  load_tile(0);

  for (int kt = 0; kt < NITER; ++kt) {
    __syncthreads();
    store_tile();
    __syncthreads();
    if (kt + 1 < NITER) load_tile(kt + 1);

    V16 ahi, alo;
    {
      const int ab = (wave * 16 + l16) * LDSTR;
      ahi.h[0] = *(const v8bf*)&sAhi[ab + kh * 8];
      ahi.h[1] = *(const v8bf*)&sAhi[ab + 16 + kh * 8];
      alo.h[0] = *(const v8bf*)&sAlo[ab + kh * 8];
      alo.h[1] = *(const v8bf*)&sAlo[ab + 16 + kh * 8];
    }

#pragma unroll
    for (int g = 0; g < 4; ++g) {
#pragma unroll
      for (int nt = 0; nt < 2; ++nt) {
        V16 bhi, blo;
        const int bb = (g * 32 + nt * 16 + l16) * LDSTR + kh * 8;
        bhi.h[0] = *(const v8bf*)&sBhi[bb];
        bhi.h[1] = *(const v8bf*)&sBhi[bb + 16];
        blo.h[0] = *(const v8bf*)&sBlo[bb];
        blo.h[1] = *(const v8bf*)&sBlo[bb + 16];

        v8f a = acc[g][nt];
        a = __builtin_amdgcn_wmma_f32_16x16x32_bf16(false, ahi.v, false, bhi.v,
                                                    (short)0, a, false, false);
        a = __builtin_amdgcn_wmma_f32_16x16x32_bf16(false, alo.v, false, bhi.v,
                                                    (short)0, a, false, false);
        a = __builtin_amdgcn_wmma_f32_16x16x32_bf16(false, ahi.v, false, blo.v,
                                                    (short)0, a, false, false);
        acc[g][nt] = a;
      }
    }
  }

  float bias[4][2];
#pragma unroll
  for (int g = 0; g < 4; ++g)
#pragma unroll
    for (int nt = 0; nt < 2; ++nt) {
      const int hc = n0 + nt * 16 + l16;
      bias[g][nt] = P.bI[g][hc] + P.bH[g][hc];
    }

#pragma unroll
  for (int nt = 0; nt < 2; ++nt) {
    const int hc = n0 + nt * 16 + l16;
#pragma unroll
    for (int r = 0; r < 8; ++r) {
      const int mrow = rowBase + wave * 16 + r + kh * 8;
      const size_t idx = (size_t)mrow * H_DIM + hc;
      const float iv = sigmoid_f(acc[0][nt][r] + bias[0][nt]);
      const float fv = sigmoid_f(acc[1][nt][r] + bias[1][nt]);
      const float gv = tanh_f   (acc[2][nt][r] + bias[2][nt]);
      const float ov = sigmoid_f(acc[3][nt][r] + bias[3][nt]);
      const float cn = fv * P.cIn[idx] + iv * gv;
      const float hn = ov * tanh_f(cn);
      P.hOut0[idx] = hn;
      if (P.hOut1) P.hOut1[idx] = hn;
      if (P.cOut)  P.cOut[idx]  = cn;
    }
  }
}

// ---------------------------------------------------------------------------
extern "C" void kernel_launch(void* const* d_in, const int* in_sizes, int n_in,
                              void* d_out, int out_size, void* d_ws, size_t ws_size,
                              hipStream_t stream) {
  (void)in_sizes; (void)n_in;

  const size_t BH    = (size_t)B_DIM * H_DIM;
  const size_t PLANE = (size_t)4096 * KW;            // elements per bf16 plane
  const size_t need  = 4 * PLANE * sizeof(__bf16);   // 64 MB workspace

  float* out = (float*)d_out;
  const int slots = (int)((size_t)out_size / BH);    // 3: (h,h,c) 2: (h,c) 1: h
  float* hOut0 = out;
  float* hOut1 = (slots >= 3) ? out + BH : nullptr;
  float* cOut  = (slots >= 2) ? out + (size_t)(slots - 1) * BH : nullptr;

  if (d_ws != nullptr && ws_size >= need) {
    __bf16* Ahi = (__bf16*)d_ws;
    __bf16* Alo = Ahi + PLANE;
    __bf16* Whi = Alo + PLANE;
    __bf16* Wlo = Whi + PLANE;

    CvtParams C;
    C.A[0] = (const float*)d_in[0];
    C.A[1] = (const float*)d_in[1];
    for (int g = 0; g < 4; ++g) {
      C.W[0][g] = (const float*)d_in[3 + 2 * g];
      C.W[1][g] = (const float*)d_in[11 + 2 * g];
    }
    C.Ahi = Ahi; C.Alo = Alo; C.Whi = Whi; C.Wlo = Wlo;
    lstm_convert_kernel<<<dim3(4096, 2), 256, 0, stream>>>(C);

    GemmParams G;
    G.Ahi = Ahi; G.Alo = Alo; G.Whi = Whi; G.Wlo = Wlo;
    for (int g = 0; g < 4; ++g) {
      G.bI[g] = (const float*)d_in[4 + 2 * g];
      G.bH[g] = (const float*)d_in[12 + 2 * g];
    }
    G.cIn = (const float*)d_in[2];
    G.hOut0 = hOut0; G.hOut1 = hOut1; G.cOut = cOut;
    lstm_wmma_async_kernel<<<dim3(H_DIM / BN, B_DIM / BM), 256, 0, stream>>>(G);
  } else {
    LstmParams P;
    P.A[0] = (const float*)d_in[0];
    P.A[1] = (const float*)d_in[1];
    P.cIn  = (const float*)d_in[2];
    for (int g = 0; g < 4; ++g) {
      P.W[0][g] = (const float*)d_in[3 + 2 * g];
      P.bI[g]   = (const float*)d_in[4 + 2 * g];
      P.W[1][g] = (const float*)d_in[11 + 2 * g];
      P.bH[g]   = (const float*)d_in[12 + 2 * g];
    }
    P.hOut0 = hOut0; P.hOut1 = hOut1; P.cOut = cOut;
    lstm_wmma_fused_kernel<<<dim3(H_DIM / BN, B_DIM / BM), 256, 0, stream>>>(P);
  }
}
